// SelfAttention_609885356558
// MI455X (gfx1250) — matrix operand order repaired and checked by reading, not compiled, not run on hardware
//
#include <hip/hip_runtime.h>
#include <hip/hip_fp16.h>

typedef _Float16 half_t;
typedef __attribute__((ext_vector_type(16))) _Float16 v16h;
typedef __attribute__((ext_vector_type(8)))  _Float16 v8h;
typedef __attribute__((ext_vector_type(8)))  float    v8f;
typedef __attribute__((ext_vector_type(4)))  unsigned int u32x4;
typedef __attribute__((ext_vector_type(8)))  unsigned int u32x8;

#define N_PIX 4096
#define C_IN  256
#define BATCH 4

// ---------------------------------------------------------------------------
// Pass 1: 1x1-conv projection as WMMA GEMM, X tiles staged by the Tensor Data
// Mover (TDM).
//   out(o,n) = sum_c W[o,c] * x[b,c,n] + bias[o]
// Block = 128 threads (4 waves). Block tile: 16 (o) x 64 (n); each wave owns a
// 16x16 output tile. K loop over C_IN in steps of 32 (one f16 WMMA per step).
// Wave 0 issues one tensor_load_to_lds per k-step: 2D tile 64x32 fp32, row
// stride N_PIX, with TDM padding (1 DWORD per 64) to produce the [32][65]
// bank-spread layout. transpose_out=1 stores [B,N,O] f16 (Q/K), else [B,O,N].
// ---------------------------------------------------------------------------
__global__ __launch_bounds__(128)
void proj_wmma_kernel(const float* __restrict__ W, const float* __restrict__ bias,
                      const float* __restrict__ x, half_t* __restrict__ out,
                      int O, int transpose_out)
{
    __shared__ __align__(16) float xs[32][65];   // 64 data + 1 TDM pad DWORD per row
    const int b    = blockIdx.z;
    const int o0   = blockIdx.y * 16;
    const int n0   = blockIdx.x * 64;
    const int tid  = threadIdx.x;
    const int wave = tid >> 5;
    const int lane = tid & 31;
    const int l16  = lane & 15;
    const int hi   = lane >> 4;

    v8f acc = {};

    // A-fragment source: W row (o0+l16); A layout K split: lanes<16 hold
    // K {0..7,16..23}, lanes>=16 hold K {8..15,24..31} (ISA 7.12.2).
    const float* wrow = W + (size_t)(o0 + l16) * C_IN + (hi ? 8 : 0);
    const int ncol = n0 + wave * 16 + l16;

    // Flat shared-aperture address truncates to the LDS byte address (ISA 10.2)
    const unsigned int lds_base = (unsigned int)(uintptr_t)&xs[0][0];

    // TDM D# group 1: geometry, invariant across the k loop (ISA 8.4)
    u32x8 g1;
    g1[0] = (2u << 16)            // data_size = 4 bytes
          | (1u << 20)            // pad_enable
          | (5u << 22);           // pad_interval = 64 DWORDs (pad_amount=0 -> +1 DWORD)
    g1[1] = (64u << 16);          // tensor_dim0 = 64 (low 16 bits in [31:16])
    g1[2] = (32u << 16);          // tensor_dim0 hi = 0 | tensor_dim1 = 32 (low 16)
    g1[3] = (64u << 16);          // tensor_dim1 hi = 0 | tile_dim0 = 64
    g1[4] = 32u;                  // tile_dim1 = 32 | tile_dim2 = 0
    g1[5] = (unsigned int)N_PIX;  // tensor_dim0_stride (low 32)
    g1[6] = 0u;                   // stride hi | tensor_dim1_stride lo (unused, 2D)
    g1[7] = 0u;
    const u32x4 g2 = {0u, 0u, 0u, 0u};   // dims 3/4 unused (2D tile)
    const u32x4 g3 = {0u, 0u, 0u, 0u};

    for (int c0 = 0; c0 < C_IN; c0 += 32) {
        __syncthreads();   // previous iteration's LDS gathers done
        if (wave == 0) {
            unsigned long long ga = (unsigned long long)(uintptr_t)
                (x + ((size_t)b * C_IN + c0) * N_PIX + n0);
            u32x4 g0;
            g0[0] = 1u;                           // count = 1 (valid descriptor)
            g0[1] = lds_base;                     // LDS byte address
            g0[2] = (unsigned int)ga;             // global_addr[31:0]
            g0[3] = ((unsigned int)(ga >> 32) & 0x01FFFFFFu)  // global_addr[56:32]
                  | (2u << 30);                   // type = 2 ("image")
            asm volatile("tensor_load_to_lds %0, %1, %2, %3"
                         :: "s"(g0), "s"(g1), "s"(g2), "s"(g3)
                         : "memory");
        }
        __builtin_amdgcn_s_wait_tensorcnt(0);     // no-op for waves 1..3
        __syncthreads();                          // tile visible to all waves

        v16h a;
#pragma unroll
        for (int j = 0; j < 8; ++j) {
            a[j]     = (half_t)wrow[c0 + j];
            a[8 + j] = (half_t)wrow[c0 + 16 + j];
        }
        // B-fragment: X tile 32x16; lanes<16 hold K {0..7,16..23}, hi lanes K {8..15,24..31}
        v16h bf;
#pragma unroll
        for (int j = 0; j < 16; ++j)
            bf[j] = (half_t)xs[(j >> 3) * 16 + hi * 8 + (j & 7)][wave * 16 + l16];

        acc = __builtin_amdgcn_wmma_f32_16x16x32_f16(false, a, false, bf,
                                                     (short)0, acc, false, false);
    }

    // C layout: VGPR r, lanes<16 -> (o0+r, ncol); hi lanes -> (o0+8+r, ncol)
#pragma unroll
    for (int r = 0; r < 8; ++r) {
        int o = o0 + r + hi * 8;
        float val = acc[r] + bias[o];
        if (transpose_out)
            out[((size_t)b * N_PIX + ncol) * O + o] = (half_t)val;
        else
            out[((size_t)b * O + o) * N_PIX + ncol] = (half_t)val;
    }
}

// ---------------------------------------------------------------------------
// Pass 2: streaming (flash) attention.
//   energy = q^T k  (K-dim = 32 = one 16x16x32 WMMA)
//   attn   = softmax_j(energy)   (online, never materialized)
//   out[b,c,i] = sum_j v[c,j]*attn[i,j] + x[b,c,i]
// Block = 128 threads (4 waves), query tile 64 rows (16 per wave); stream key
// tiles of 64. Each wave keeps the full 16x256 fp32 output accumulator in
// VGPRs (16 x v8f). Probabilities round-trip through a per-wave LDS buffer to
// re-layout C-fragment -> A-fragment for the P x V^T WMMAs.
// ---------------------------------------------------------------------------
__global__ __launch_bounds__(128)
void attn_wmma_kernel(const half_t* __restrict__ qT, const half_t* __restrict__ kT,
                      const half_t* __restrict__ v, const float* __restrict__ x,
                      float* __restrict__ out)
{
    __shared__ __align__(32) half_t pbuf[4][16][64];   // per-wave P tile, 8 KB
    const int b    = blockIdx.y;
    const int i0   = blockIdx.x * 64;
    const int tid  = threadIdx.x;
    const int wave = tid >> 5;
    const int lane = tid & 31;
    const int l16  = lane & 15;
    const int hi   = lane >> 4;
    const int ko8  = hi ? 8 : 0;    // unused; A-fragments also split K by ko16
    const int ko16 = hi ? 16 : 0;   // B-fragment K split offset

    // Q A-fragment, fixed for the whole key loop (rows i0+wave*16+l16)
    const half_t* qp = qT + ((size_t)b * N_PIX + (i0 + wave * 16 + l16)) * 32;
    v8h q0 = *(const v8h*)(qp + ko16);
    v8h q1 = *(const v8h*)(qp + ko16 + 8);
    v16h qa;
#pragma unroll
    for (int j = 0; j < 8; ++j) { qa[j] = q0[j]; qa[8 + j] = q1[j]; }

    const v8f zero = {};
    v8f m, lsum;
#pragma unroll
    for (int j = 0; j < 8; ++j) { m[j] = -__builtin_inff(); lsum[j] = 0.0f; }
    v8f acc[16];
#pragma unroll
    for (int t = 0; t < 16; ++t) acc[t] = zero;

    for (int j0 = 0; j0 < N_PIX; j0 += 64) {
        if (j0 + 64 < N_PIX) {   // warm L0/L2 for the next key tile
            __builtin_prefetch(kT + ((size_t)b * N_PIX + (j0 + 64 + lane)) * 32, 0, 1);
        }

        // ---- scores: 16 rows x 64 cols, one WMMA per 16-col sub-tile ----
        v8f s[4];
#pragma unroll
        for (int sub = 0; sub < 4; ++sub) {
            const half_t* kp = kT + ((size_t)b * N_PIX + (j0 + sub * 16 + l16)) * 32 + ko16;
            v16h kb = *(const v16h*)kp;
            s[sub] = __builtin_amdgcn_wmma_f32_16x16x32_f16(false, qa, false, kb,
                                                            (short)0, zero, false, false);
        }

        // ---- online softmax: row stats live replicated across 16-lane halves ----
        v8f mnew = m;
#pragma unroll
        for (int sub = 0; sub < 4; ++sub)
#pragma unroll
            for (int j = 0; j < 8; ++j) mnew[j] = fmaxf(mnew[j], s[sub][j]);
#pragma unroll
        for (int off = 1; off < 16; off <<= 1)
#pragma unroll
            for (int j = 0; j < 8; ++j)
                mnew[j] = fmaxf(mnew[j], __shfl_xor(mnew[j], off, 16));

        v8f alpha;
#pragma unroll
        for (int j = 0; j < 8; ++j) alpha[j] = __expf(m[j] - mnew[j]);
        m = mnew;

        v8f rsum = {};
#pragma unroll
        for (int sub = 0; sub < 4; ++sub)
#pragma unroll
            for (int j = 0; j < 8; ++j) {
                float p = __expf(s[sub][j] - mnew[j]);
                rsum[j] += p;
                // C layout -> LDS row-major: VGPR j, lanes<16 -> row j; hi -> row 8+j
                pbuf[wave][j + hi * 8][sub * 16 + l16] = (half_t)p;
            }
#pragma unroll
        for (int off = 1; off < 16; off <<= 1)
#pragma unroll
            for (int j = 0; j < 8; ++j) rsum[j] += __shfl_xor(rsum[j], off, 16);
#pragma unroll
        for (int j = 0; j < 8; ++j) lsum[j] = lsum[j] * alpha[j] + rsum[j];
#pragma unroll
        for (int t = 0; t < 16; ++t)
#pragma unroll
            for (int j = 0; j < 8; ++j) acc[t][j] *= alpha[j];

        __syncthreads();

        // ---- re-read P as A-fragments (rows = l16, K = j within tile) ----
        const half_t* pp = &pbuf[wave][0][0];
        v8h p00 = *(const v8h*)(pp + l16 * 64 + ko16);
        v8h p01 = *(const v8h*)(pp + l16 * 64 + ko16 + 8);
        v8h p10 = *(const v8h*)(pp + l16 * 64 + 32 + ko16);
        v8h p11 = *(const v8h*)(pp + l16 * 64 + 32 + ko16 + 8);
        v16h pa0, pa1;
#pragma unroll
        for (int j = 0; j < 8; ++j) {
            pa0[j] = p00[j]; pa0[8 + j] = p01[j];
            pa1[j] = p10[j]; pa1[8 + j] = p11[j];
        }

        // ---- O += P x V^T over 16 channel tiles (2 WMMAs each) ----
#pragma unroll
        for (int t = 0; t < 16; ++t) {
            const half_t* vp = v + ((size_t)b * C_IN + (t * 16 + l16)) * N_PIX + j0;
            v16h vb0 = *(const v16h*)(vp + ko16);          // j = j0 .. j0+31
            v16h vb1 = *(const v16h*)(vp + 32 + ko16);     // j = j0+32 .. j0+63
            acc[t] = __builtin_amdgcn_wmma_f32_16x16x32_f16(false, pa0, false, vb0,
                                                            (short)0, acc[t], false, false);
            acc[t] = __builtin_amdgcn_wmma_f32_16x16x32_f16(false, pa1, false, vb1,
                                                            (short)0, acc[t], false, false);
        }
        __syncthreads();
    }

    // normalize and write out with residual (+x)
    v8f inv;
#pragma unroll
    for (int j = 0; j < 8; ++j) inv[j] = 1.0f / lsum[j];

#pragma unroll
    for (int t = 0; t < 16; ++t) {
#pragma unroll
        for (int r = 0; r < 8; ++r) {
            int crow = t * 16 + l16;
            int irow = i0 + wave * 16 + r + hi * 8;
            size_t idx = ((size_t)b * C_IN + crow) * N_PIX + irow;
            out[idx] = acc[t][r] * inv[r] + x[idx];
        }
    }
}

// ---------------------------------------------------------------------------
extern "C" void kernel_launch(void* const* d_in, const int* in_sizes, int n_in,
                              void* d_out, int out_size, void* d_ws, size_t ws_size,
                              hipStream_t stream)
{
    (void)in_sizes; (void)n_in; (void)out_size; (void)ws_size;
    const float* x  = (const float*)d_in[0];
    const float* Wq = (const float*)d_in[1];
    const float* bq = (const float*)d_in[2];
    const float* Wk = (const float*)d_in[3];
    const float* bk = (const float*)d_in[4];
    const float* Wv = (const float*)d_in[5];
    const float* bv = (const float*)d_in[6];
    float* out = (float*)d_out;

    // f16 workspace: qT [B,N,32] (1 MB) | kT [B,N,32] (1 MB) | v [B,C,N] (8 MB)
    half_t* qT = (half_t*)d_ws;
    half_t* kT = qT + (size_t)BATCH * N_PIX * 32;
    half_t* vv = kT + (size_t)BATCH * N_PIX * 32;

    dim3 blk(128);
    proj_wmma_kernel<<<dim3(N_PIX / 64,  2, BATCH), blk, 0, stream>>>(Wq, bq, x, qT,  32, 1);
    proj_wmma_kernel<<<dim3(N_PIX / 64,  2, BATCH), blk, 0, stream>>>(Wk, bk, x, kT,  32, 1);
    proj_wmma_kernel<<<dim3(N_PIX / 64, 16, BATCH), blk, 0, stream>>>(Wv, bv, x, vv, 256, 0);
    attn_wmma_kernel<<<dim3(N_PIX / 64, BATCH), blk, 0, stream>>>(qT, kT, vv, x, out);
}
